// VariationalGNN_56504589746551
// MI455X (gfx1250) — matrix operand order repaired and checked by reading, not compiled, not run on hardware
//
#include <hip/hip_runtime.h>
#include <hip/hip_bf16.h>
#include <math.h>

// ---------------------------------------------------------------------------
// VariationalGNN, dead-code-eliminated to the 513 obs_out nodes.
// All dense 256-wide GEMMs run on the CDNA5 matrix pipe via
// v_wmma_f32_16x16x32_f16 (f16 inputs, f32 accumulate).
// ---------------------------------------------------------------------------

#define FEAT    256
#define NSEL    513      // |obs| + last node
#define NOBS    512      // |obs|
#define MPAD    528      // NSEL padded to a multiple of 16 (33 tiles)
#define LRALPHA 0.2f
#define INV_DH  0.0625f  // 1/sqrt(FEAT) = 1/16
#define LN_EPS  1e-6f

typedef __attribute__((ext_vector_type(16))) _Float16 v16h;
typedef __attribute__((ext_vector_type(8)))  float    v8f;

__device__ __forceinline__ float block_sum_256(float v, float* red) {
  const int t = threadIdx.x;
  red[t] = v; __syncthreads();
  #pragma unroll
  for (int s = 128; s > 0; s >>= 1) {
    if (t < s) red[t] += red[t + s];
    __syncthreads();
  }
  float r = red[0];
  __syncthreads();
  return r;
}

// --------------------- gather selected node embeddings ---------------------
// output_edges row0[0:513] == obs_out (tile() puts the sorted node list first)
__global__ void k_gather(const int* __restrict__ out_edges,
                         const float* __restrict__ embed,
                         float* __restrict__ h) {
  const int i = blockIdx.x;   // 0..MPAD-1
  const int f = threadIdx.x;  // 0..255
  float v = 0.0f;
  if (i < NSEL) {
    const int node = out_edges[i];
    v = embed[(size_t)node * FEAT + f];
  }
  h[(size_t)i * FEAT + f] = v;
}

// --------------------- WMMA GEMM: C = A(MxK) * op(B) + bias ----------------
// BT=true : B is (N x K) row-major -> C = A @ B^T   (x @ W.T case)
// BT=false: B is (K x N) row-major -> C = A @ B     (E @ D case)
template <bool BT>
__global__ void k_gemm_wmma(const float* __restrict__ A,
                            const float* __restrict__ B,
                            const float* __restrict__ bias,
                            float* __restrict__ C,
                            int M, int K, int N) {
  const int tm   = blockIdx.x;
  const int tn   = blockIdx.y;
  const int lane = threadIdx.x;   // 0..31 (wave32)
  const int half = lane >> 4;     // 0 / 1
  const int l16  = lane & 15;

  const int m_a = tm * 16 + l16;  // A row this lane supplies
  const int n_b = tn * 16 + l16;  // output column owned by this lane
  const float* Arow = A + (size_t)m_a * K;

  v8f acc = {};
  for (int kk = 0; kk < K; kk += 32) {
    // A 16x32 f16 fragment: lanes 0-15 K=0..15, lanes 16-31 K=16..31
    v16h a;
    const int ka0 = kk + half * 16;
    const int ka1 = kk + half * 16 + 8;
    #pragma unroll
    for (int e = 0; e < 8; ++e) {
      a[e]     = (_Float16)Arow[ka0 + e];
      a[e + 8] = (_Float16)Arow[ka1 + e];
    }
    // B 32x16 f16 fragment: lanes 0-15 hold K=kk..kk+15, lanes 16-31 K=kk+16..kk+31
    v16h b;
    if (BT) {
      const float* Brow = B + (size_t)n_b * K + kk + half * 16;  // contiguous
      #pragma unroll
      for (int e = 0; e < 16; ++e) b[e] = (_Float16)Brow[e];
    } else {
      const float* Bp = B + (size_t)(kk + half * 16) * N + n_b;  // stride N
      #pragma unroll
      for (int e = 0; e < 16; ++e) b[e] = (_Float16)Bp[(size_t)e * N];
    }
    acc = __builtin_amdgcn_wmma_f32_16x16x32_f16(false, a, false, b,
                                                 (short)0, acc, false, false);
  }
  const float bv = bias ? bias[n_b] : 0.0f;
  #pragma unroll
  for (int e = 0; e < 8; ++e) {
    const int m = tm * 16 + half * 8 + e;   // C row layout (lanes 16-31: M+8)
    C[(size_t)m * N + n_b] = acc[e] + bv;
  }
}

// --------------------- attention scores s_i, t_i ---------------------------
__global__ void k_scores(const float* __restrict__ d, const float* __restrict__ avec,
                         float* __restrict__ sn, float* __restrict__ tn) {
  __shared__ float red[256];
  const int i = blockIdx.x;   // 0..NSEL-1
  const int t = threadIdx.x;
  const float x   = d[(size_t)i * FEAT + t];
  const float ssum = block_sum_256(x * avec[t], red);
  const float tsum = block_sum_256(x * avec[FEAT + t], red);
  if (t == 0) { sn[i] = ssum; tn[i] = tsum; }
}

// --------------------- dense edge weights + rowsums ------------------------
__global__ void k_edge(const float* __restrict__ sn, const float* __restrict__ tn,
                       float* __restrict__ E, float* __restrict__ rsum) {
  __shared__ float red[256];
  const int i = blockIdx.x;   // 0..NOBS-1
  const int t = threadIdx.x;
  const float si = sn[i];
  float part = 0.0f;
  for (int j = t; j < NOBS; j += 256) {
    float sc = si + tn[j];
    sc = (sc > 0.0f) ? sc : LRALPHA * sc;   // leaky_relu(.., 0.2)
    float e = expf(sc * INV_DH);
    E[(size_t)i * NOBS + j] = e;
    part += e;
  }
  const float s = block_sum_256(part, red);
  if (t == 0) rsum[i] = s;
}

// --------------------- softmax-normalize + LayerNorm + ELU -----------------
__global__ void k_post_attn(const float* __restrict__ agg, const float* __restrict__ rsum,
                            const float* __restrict__ d,
                            const float* __restrict__ g, const float* __restrict__ b,
                            float* __restrict__ hout) {
  __shared__ float red[256];
  const int i = blockIdx.x;   // 0..MPAD-1
  const int t = threadIdx.x;
  if (i >= NSEL) { hout[(size_t)i * FEAT + t] = 0.0f; return; }  // keep pads zero
  float x = (i < NOBS) ? agg[(size_t)i * FEAT + t] / rsum[i]
                       : d[(size_t)i * FEAT + t];   // last node: rowsum==0 path
  const float mean = block_sum_256(x, red) * (1.0f / FEAT);
  const float dx   = x - mean;
  const float var  = block_sum_256(dx * dx, red) * (1.0f / (FEAT - 1)); // ddof=1
  float y = g[t] * dx / (sqrtf(var) + LN_EPS) + b[t];
  hout[(size_t)i * FEAT + t] = (y > 0.0f) ? y : (expf(y) - 1.0f);       // ELU
}

// --------------------- final head: only node 512's attention row -----------
__global__ void k_final(const float* __restrict__ sn, const float* __restrict__ tn,
                        const float* __restrict__ dO,
                        const float* __restrict__ lng, const float* __restrict__ lnb,
                        const float* __restrict__ linw, const float* __restrict__ linb,
                        const float* __restrict__ o1w, const float* __restrict__ o1b,
                        const float* __restrict__ o2w, const float* __restrict__ o2b,
                        float* __restrict__ out) {
  __shared__ float red[256];
  __shared__ float eSh[NSEL];
  __shared__ float vec[256];
  const int t = threadIdx.x;

  const float s_last = sn[NSEL - 1];
  float part = 0.0f;
  for (int j = t; j < NSEL; j += 256) {
    float sc = s_last + tn[j];
    sc = (sc > 0.0f) ? sc : LRALPHA * sc;
    float e = expf(sc * INV_DH);
    eSh[j] = e;
    part += e;
  }
  const float rowsum = block_sum_256(part, red);  // sync also publishes eSh

  float acc = 0.0f;
  for (int j = 0; j < NSEL; ++j) acc += eSh[j] * dO[(size_t)j * FEAT + t];
  float x = acc / rowsum;

  const float mean = block_sum_256(x, red) * (1.0f / FEAT);
  const float dx   = x - mean;
  const float var  = block_sum_256(dx * dx, red) * (1.0f / (FEAT - 1));
  float y = lng[t] * dx / (sqrtf(var) + LN_EPS) + lnb[t];
  y = fmaxf(y, 0.0f);                     // relu(layernorm(g))
  vec[t] = y; __syncthreads();

  float gl = linb[t];                     // g = y @ lin_w.T + lin_b
  for (int f = 0; f < FEAT; ++f) gl += vec[f] * linw[(size_t)t * FEAT + f];
  float z = fmaxf(gl, 0.0f);              // z = relu(g[-1])
  __syncthreads();
  vec[t] = z; __syncthreads();

  float z2 = o1b[t];                      // relu(z @ out1_w.T + out1_b)
  for (int f = 0; f < FEAT; ++f) z2 += vec[f] * o1w[(size_t)t * FEAT + f];
  z2 = fmaxf(z2, 0.0f);

  const float lp = block_sum_256(z2 * o2w[t], red);
  if (t == 0) out[0] = lp + o2b[0];
}

// ---------------------------------------------------------------------------
extern "C" void kernel_launch(void* const* d_in, const int* in_sizes, int n_in,
                              void* d_out, int out_size, void* d_ws, size_t ws_size,
                              hipStream_t stream) {
  (void)in_sizes; (void)n_in; (void)out_size; (void)ws_size;
  const float* embed   = (const float*)d_in[0];
  const float* W_w     = (const float*)d_in[1];   // (2,256,256)
  const float* W_b     = (const float*)d_in[2];   // (2,256)
  const float* a_vec   = (const float*)d_in[3];   // (2,512)
  const float* ln_a    = (const float*)d_in[4];
  const float* ln_b    = (const float*)d_in[5];
  const float* Wo_w    = (const float*)d_in[6];
  const float* Wo_b    = (const float*)d_in[7];
  const float* ao      = (const float*)d_in[8];
  const float* lno_a   = (const float*)d_in[9];
  const float* lno_b   = (const float*)d_in[10];
  const float* lin_w   = (const float*)d_in[11];
  const float* lin_b   = (const float*)d_in[12];
  const float* param_w = (const float*)d_in[13];  // only first 256 rows used
  const float* param_b = (const float*)d_in[14];
  const float* out1_w  = (const float*)d_in[15];
  const float* out1_b  = (const float*)d_in[16];
  const float* out2_w  = (const float*)d_in[17];
  const float* out2_b  = (const float*)d_in[18];
  const int*   out_edges = (const int*)d_in[20];  // row0[0:513] = obs_out

  float* ws   = (float*)d_ws;
  float* h    = ws;                        // MPAD*FEAT
  float* dbuf = h    + (size_t)MPAD * FEAT;
  float* agg  = dbuf + (size_t)MPAD * FEAT;
  float* E    = agg  + (size_t)MPAD * FEAT; // NOBS*NOBS
  float* rsum = E    + (size_t)NOBS * NOBS;
  float* sn   = rsum + NOBS;
  float* tn   = sn   + MPAD;

  const dim3 g33(MPAD / 16, FEAT / 16);   // 33 x 16 tiles
  const dim3 g32(NOBS / 16, FEAT / 16);   // 32 x 16 tiles

  k_gather<<<MPAD, FEAT, 0, stream>>>(out_edges, embed, h);

  for (int l = 0; l < 2; ++l) {
    k_gemm_wmma<true ><<<g33, 32, 0, stream>>>(h, W_w + (size_t)l * FEAT * FEAT,
                                               W_b + l * FEAT, dbuf, MPAD, FEAT, FEAT);
    k_scores<<<NSEL, FEAT, 0, stream>>>(dbuf, a_vec + l * 2 * FEAT, sn, tn);
    k_edge<<<NOBS, FEAT, 0, stream>>>(sn, tn, E, rsum);
    k_gemm_wmma<false><<<g32, 32, 0, stream>>>(E, dbuf, nullptr, agg, NOBS, NOBS, FEAT);
    k_post_attn<<<MPAD, FEAT, 0, stream>>>(agg, rsum, dbuf,
                                           ln_a + l * FEAT, ln_b + l * FEAT, h);
  }

  // hp[:,0,:] = h @ param_w[:256].T + param_b[:256]
  k_gemm_wmma<true><<<g33, 32, 0, stream>>>(h, param_w, param_b, dbuf, MPAD, FEAT, FEAT);
  // output-attention pre-projection: d_o = h' @ Wo_w.T + Wo_b
  k_gemm_wmma<true><<<g33, 32, 0, stream>>>(dbuf, Wo_w, Wo_b, agg, MPAD, FEAT, FEAT);
  k_scores<<<NSEL, FEAT, 0, stream>>>(agg, ao, sn, tn);
  k_final<<<1, FEAT, 0, stream>>>(sn, tn, agg, lno_a, lno_b, lin_w, lin_b,
                                  out1_w, out1_b, out2_w, out2_b, (float*)d_out);
}
